// node_bottle_net_21534966022302
// MI455X (gfx1250) — matrix operand order repaired and checked by reading, not compiled, not run on hardware
//
#include <hip/hip_runtime.h>
#include <hip/hip_bf16.h>

// ---------------------------------------------------------------------------
// node_bottle_net on MI455X (gfx1250, wave32)
//   emb = ELU(x @ (W1@W0)^T + (W1@b0 + b1))   -- linears fused (no act between)
//   out[d] += emb[s] over 3.2M edges           -- L2-resident scatter
// ---------------------------------------------------------------------------

#define IN_DIM  256
#define HID_DIM 256
#define OUT_DIM 128

typedef __attribute__((ext_vector_type(16))) __bf16 v16bf;
typedef __attribute__((ext_vector_type(8)))  float  v8f;
typedef __attribute__((ext_vector_type(4)))  int    v4i;

union BFrag {
    uint4 u[2];
    v16bf v;
};

// Optional CDNA5 async global->LDS path (probe via __has_builtin so the file
// compiles on any toolchain; host pass sees the fallback).
#define USE_ASYNC_LDS 0
#if defined(__has_builtin)
#if __has_builtin(__builtin_amdgcn_global_load_async_to_lds_b128) && \
    __has_builtin(__builtin_amdgcn_s_wait_asynccnt)
#undef USE_ASYNC_LDS
#define USE_ASYNC_LDS 1
#define AS1 __attribute__((address_space(1)))
#define AS3 __attribute__((address_space(3)))
#endif
#endif

// ---------------------------------------------------------------------------
// Kernel 1: fold the two linears into one [OUT_DIM, IN_DIM] bf16 weight and an
// f32 bias.  Wf[o][i] = sum_h W1[o][h] * W0[h][i];  bf[o] = W1[o]·b0 + b1[o].
// ---------------------------------------------------------------------------
__global__ __launch_bounds__(256)
void fuse_weights_kernel(const float* __restrict__ W0, const float* __restrict__ b0,
                         const float* __restrict__ W1, const float* __restrict__ b1,
                         __bf16* __restrict__ Wf, float* __restrict__ bf)
{
    __shared__ float red[256];
    const int o = blockIdx.x;    // 0..127
    const int i = threadIdx.x;   // 0..255
    const float* __restrict__ w1row = W1 + o * HID_DIM;

    float acc = 0.0f;
    #pragma unroll 8
    for (int h = 0; h < HID_DIM; ++h)
        acc = fmaf(w1row[h], W0[h * IN_DIM + i], acc);   // W0 col read coalesced
    Wf[o * IN_DIM + i] = (__bf16)acc;

    // fused bias via block reduction
    red[i] = w1row[i] * b0[i];
    __syncthreads();
    for (int s = 128; s > 0; s >>= 1) {
        if (i < s) red[i] += red[i + s];
        __syncthreads();
    }
    if (i == 0) bf[o] = red[0] + b1[o];
}

// ---------------------------------------------------------------------------
// Kernel 2: zero-fill output accumulator (harness poisons d_out with 0xAA).
// ---------------------------------------------------------------------------
__global__ __launch_bounds__(256)
void zero_out_kernel(float4* __restrict__ out4, int n4)
{
    const float4 z = make_float4(0.f, 0.f, 0.f, 0.f);
    for (int idx = blockIdx.x * blockDim.x + threadIdx.x; idx < n4;
         idx += gridDim.x * blockDim.x)
        out4[idx] = z;
}

// ---------------------------------------------------------------------------
// Kernel 3: fused GEMM + bias + ELU using v_wmma_f32_16x16x32_bf16.
//   Block = 256 threads = 8 waves, tile = 128 nodes x 128 outputs.
//   Wave w owns nodes [base + 16w, +16) and all 8 output 16-col tiles.
//   Weights (128x256 bf16 = 64KB) staged in LDS once per block (async TDM-lite
//   path: global_load_async_to_lds_b128 + s_wait_asynccnt).
//   Per K-step: batch-load all 8 B fragments, then 8 back-to-back WMMAs.
//   Epilogue: bounce tile through per-wave LDS slice -> coalesced b128 stores.
// ---------------------------------------------------------------------------
__global__ __launch_bounds__(256)
void gemm_elu_kernel(const float* __restrict__ x, const __bf16* __restrict__ Wf,
                     const float* __restrict__ bf, float* __restrict__ emb,
                     int n_nodes)
{
    __shared__ __align__(16) __bf16 ldsW[OUT_DIM * IN_DIM];   // 64 KB weights
    __shared__ __align__(16) float  ldsOut[128 * OUT_DIM];    // 64 KB epilogue

    // --- stage fused weights into LDS ---
#if USE_ASYNC_LDS
    {
        uint4* gsrc = (uint4*)Wf;
        uint4* ldst = (uint4*)ldsW;
        #pragma unroll
        for (int k = 0; k < (OUT_DIM * IN_DIM) / 8 / 256; ++k) {
            const int idx = threadIdx.x + k * 256;
            __builtin_amdgcn_global_load_async_to_lds_b128(
                (AS1 v4i*)(unsigned long long)(gsrc + idx),
                (AS3 v4i*)(unsigned int)(unsigned long long)(ldst + idx),
                0, 0);
        }
        __builtin_amdgcn_s_wait_asynccnt(0);
    }
#else
    {
        const uint4* __restrict__ src = (const uint4*)Wf;
        uint4* dst = (uint4*)ldsW;
        #pragma unroll
        for (int k = 0; k < (OUT_DIM * IN_DIM) / 8 / 256; ++k)
            dst[threadIdx.x + k * 256] = src[threadIdx.x + k * 256];
    }
#endif
    __syncthreads();

    const int wave   = threadIdx.x >> 5;
    const int lane   = threadIdx.x & 31;
    const int laneLo = lane & 15;
    const bool hiHalf = lane >= 16;

    const int nodeBase = blockIdx.x * 128 + wave * 16;

    // A-fragment source row (clamped so WMMA runs with full EXEC on the tail)
    int rowA = nodeBase + laneLo;
    if (rowA > n_nodes - 1) rowA = n_nodes - 1;
    const float* __restrict__ xrow = x + (size_t)rowA * IN_DIM;

    v8f acc[8] = {};   // 8 output tiles x 8 VGPRs

    #pragma unroll
    for (int s = 0; s < IN_DIM / 32; ++s) {        // 8 K-steps of 32
        const int k0 = s * 32;

        // prefetch next K-chunk of this row (emits global_prefetch_b8)
        if (s + 1 < IN_DIM / 32)
            __builtin_prefetch(xrow + k0 + 32, 0, 1);

        // ---- A fragment: contiguous k order (k = 16h + s), same as B ----
        // lanes 0-15 : K = k0+{0..15}
        // lanes 16-31: K = k0+{16..31}
        const int c1 = k0 + (hiHalf ? 16 : 0);
        const float4 f0 = *(const float4*)(xrow + c1);
        const float4 f1 = *(const float4*)(xrow + c1 + 4);
        const float4 f2 = *(const float4*)(xrow + c1 + 8);
        const float4 f3 = *(const float4*)(xrow + c1 + 12);

        v16bf a;
        a[0]  = (__bf16)f0.x; a[1]  = (__bf16)f0.y; a[2]  = (__bf16)f0.z; a[3]  = (__bf16)f0.w;
        a[4]  = (__bf16)f1.x; a[5]  = (__bf16)f1.y; a[6]  = (__bf16)f1.z; a[7]  = (__bf16)f1.w;
        a[8]  = (__bf16)f2.x; a[9]  = (__bf16)f2.y; a[10] = (__bf16)f2.z; a[11] = (__bf16)f2.w;
        a[12] = (__bf16)f3.x; a[13] = (__bf16)f3.y; a[14] = (__bf16)f3.z; a[15] = (__bf16)f3.w;

        // ---- batch-load ALL 8 B fragments (one dscnt wait total) ----
        // lane n<16 holds K=k0..k0+15 of column o=16t+n; lanes>=16 K=k0+16..31
        // -> 32 contiguous bytes of the row-major Wf row; two ds_load_b128.
        const int kb = k0 + (hiHalf ? 16 : 0);
        BFrag b[8];
        #pragma unroll
        for (int t = 0; t < 8; ++t) {
            const uint4* bp = (const uint4*)(ldsW + (t * 16 + laneLo) * IN_DIM + kb);
            b[t].u[0] = bp[0];
            b[t].u[1] = bp[1];
        }

        // ---- 8 back-to-back WMMAs on the matrix pipe ----
        #pragma unroll
        for (int t = 0; t < 8; ++t)
            acc[t] = __builtin_amdgcn_wmma_f32_16x16x32_bf16(
                         /*neg_a=*/false, a, /*neg_b=*/false, b[t].v,
                         /*c_mod=*/(short)0, acc[t],
                         /*reuse_a=*/false, /*reuse_b=*/false);
    }

    // ---- epilogue: bias + ELU -> per-wave LDS slice -> coalesced stores ----
    // C/D layout: lane l<16 -> N=l, VGPR r -> M=r ; lanes 16-31 -> N=l-16, M=r+8
    // Each wave writes/reads only its own 8KB slice: no cross-wave barrier
    // needed (per-wave DS ordering + dscnt wait handled by the compiler).
    float* myOut = ldsOut + wave * (16 * OUT_DIM);
    #pragma unroll
    for (int t = 0; t < 8; ++t) {
        const int o = t * 16 + laneLo;
        const float bias = bf[o];
        #pragma unroll
        for (int r = 0; r < 8; ++r) {
            float v = acc[t][r] + bias;
            v = (v > 0.0f) ? v : (__expf(v) - 1.0f);   // ELU, alpha=1
            myOut[(r + (hiHalf ? 8 : 0)) * OUT_DIM + o] = v;
        }
    }
    // write back: 16 rows x 512B, each row one fully-coalesced b128 per lane
    #pragma unroll
    for (int r = 0; r < 16; ++r) {
        const int node = nodeBase + r;
        if (node < n_nodes) {
            const float4 v = *(const float4*)(myOut + r * OUT_DIM + lane * 4);
            *(float4*)(emb + (size_t)node * OUT_DIM + lane * 4) = v;
        }
    }
}

// ---------------------------------------------------------------------------
// Kernel 4: edge scatter  out[dst[e]] += emb[src[e]].
//   One wave32 per edge, one float4 per lane = 128 channels.
//   emb(51MB)+out(51MB)+indices(25MB) fit in the 192MB L2 -> L2-resident.
// ---------------------------------------------------------------------------
__global__ __launch_bounds__(256)
void scatter_sum_kernel(const float* __restrict__ emb, const int* __restrict__ src,
                        const int* __restrict__ dst, float* __restrict__ out,
                        int n_edges)
{
    const int lane   = threadIdx.x & 31;
    const int wave   = (blockIdx.x * blockDim.x + threadIdx.x) >> 5;
    const int nWaves = (gridDim.x * blockDim.x) >> 5;

    for (int e = wave; e < n_edges; e += nWaves) {
        const int s = src[e];   // uniform within wave -> broadcast from L0
        const int d = dst[e];
        const float4 v = *(const float4*)(emb + (size_t)s * OUT_DIM + lane * 4);
        float* op = out + (size_t)d * OUT_DIM + lane * 4;
        atomicAdd(op + 0, v.x);
        atomicAdd(op + 1, v.y);
        atomicAdd(op + 2, v.z);
        atomicAdd(op + 3, v.w);
    }
}

// ---------------------------------------------------------------------------
// Launch
// ---------------------------------------------------------------------------
extern "C" void kernel_launch(void* const* d_in, const int* in_sizes, int n_in,
                              void* d_out, int out_size, void* d_ws, size_t ws_size,
                              hipStream_t stream)
{
    const float* x  = (const float*)d_in[0];   // [N, 256]
    const float* W0 = (const float*)d_in[1];   // [256, 256]
    const float* b0 = (const float*)d_in[2];   // [256]
    const float* W1 = (const float*)d_in[3];   // [128, 256]
    const float* b1 = (const float*)d_in[4];   // [128]
    const int*  src = (const int*)d_in[5];     // [E]
    const int*  dst = (const int*)d_in[6];     // [E]
    float*      out = (float*)d_out;           // [N, 128]

    const int n_nodes = in_sizes[0] / IN_DIM;
    const int n_edges = in_sizes[5];

    // workspace layout
    char* ws = (char*)d_ws;
    const size_t EMB_BYTES = (size_t)n_nodes * OUT_DIM * sizeof(float);
    float*  emb = (float*)ws;
    __bf16* Wf  = (__bf16*)(ws + EMB_BYTES);
    float*  bf  = (float*)(ws + EMB_BYTES + (size_t)OUT_DIM * IN_DIM * sizeof(__bf16));

    // 1) fold the two linears (no activation between them in the reference)
    fuse_weights_kernel<<<OUT_DIM, 256, 0, stream>>>(W0, b0, W1, b1, Wf, bf);

    // 2) zero output accumulator
    {
        const int n4 = out_size / 4;
        const int blocks = (n4 + 255) / 256 < 4096 ? (n4 + 255) / 256 : 4096;
        zero_out_kernel<<<blocks, 256, 0, stream>>>((float4*)out, n4);
    }

    // 3) fused GEMM + bias + ELU (bf16 WMMA, f32 accumulate)
    {
        const int blocks = (n_nodes + 127) / 128;
        gemm_elu_kernel<<<blocks, 256, 0, stream>>>(x, Wf, bf, emb, n_nodes);
    }

    // 4) edge scatter-sum (wave per edge)
    scatter_sum_kernel<<<4096, 256, 0, stream>>>(emb, src, dst, out, n_edges);
}
